// TopKPoolBroadcastGCN_21964462752258
// MI455X (gfx1250) — matrix operand order repaired and checked by reading, not compiled, not run on hardware
//
#include <hip/hip_runtime.h>
#include <hip/hip_bf16.h>

// ---------------------------------------------------------------------------
// TopK-pool GCN pipeline for gfx1250 (wave32, WMMA, TDM).
// GEMMs: v_wmma_f32_16x16x32_bf16, 2 column tiles per wave (A-fragment reuse).
// x@W1 stages W1 into LDS via tensor_load_to_lds (TENSORcnt) when available.
// Scatter/aggregate paths stay fp32 atomics (bandwidth-bound at 23.3 TB/s).
// ---------------------------------------------------------------------------

constexpr int cN   = 100000;
constexpr int cE   = 1600000;
constexpr int cIN  = 128;
constexpr int cHID = 128;
constexpr int cOUT = 64;
constexpr int cK   = 1024;

typedef __attribute__((ext_vector_type(16))) __bf16 bf16x16;
typedef __attribute__((ext_vector_type(8)))  float  f32x8;
typedef __attribute__((ext_vector_type(4)))  unsigned int u32x4;
typedef __attribute__((ext_vector_type(8)))  int i32x8;
typedef __attribute__((ext_vector_type(4)))  int i32x4;

#if defined(__has_builtin)
#  if __has_builtin(__builtin_amdgcn_tensor_load_to_lds)
#    define CDNA5_HAS_TDM 1
#  endif
#endif
#ifndef CDNA5_HAS_TDM
#  define CDNA5_HAS_TDM 0
#endif
#if defined(__has_include)
#  if __has_include(<hip/amd_detail/amd_gfx1250_TDM.h>)
#    define CDNA5_TDM_6ARG 1
#  endif
#endif
#ifndef CDNA5_TDM_6ARG
#  define CDNA5_TDM_6ARG 0
#endif

struct Ctl {
  unsigned prefix, mask, need, T;
  unsigned gtCount, tieCount, keptCount, jmax;
  unsigned hist[256];
};

#define NB(n, t) (((n) + (t) - 1) / (t))

// ----------------------------- helpers -------------------------------------

__device__ __forceinline__ unsigned ordkey(float f) {
  unsigned u = __float_as_uint(f);
  return (u & 0x80000000u) ? ~u : (u | 0x80000000u);
}

// ISA 7.12.2 wave32 fragment layouts (16-bit A 16x32; B 32x16; C/D f32 16x16).
__device__ __forceinline__ bf16x16 load_a_frag(const float* __restrict__ A,
                                               size_t a_rs, size_t a_cs,
                                               int m0, int kk, int half, int q) {
  bf16x16 a;
#pragma unroll
  for (int e = 0; e < 16; ++e) {
    int v = e >> 1;
    int k = ((v < 4) ? (v << 1) : (16 + ((v - 4) << 1))) + (half << 3) + (e & 1);
    a[e] = (__bf16)A[(size_t)(m0 + q) * a_rs + (size_t)(kk + k) * a_cs];
  }
  return a;
}

__device__ __forceinline__ bf16x16 load_b_frag(const float* __restrict__ B,
                                               size_t b_rs, size_t b_cs,
                                               int n0, int kk, int half, int q) {
  bf16x16 b;
#pragma unroll
  for (int e = 0; e < 16; ++e) {
    int k = ((e < 8) ? e : (e + 8)) + (half << 3);
    b[e] = (__bf16)B[(size_t)(kk + k) * b_rs + (size_t)(n0 + q) * b_cs];
  }
  return b;
}

#define WMMA_BF16(a, b, c)                                                     \
  __builtin_amdgcn_wmma_f32_16x16x32_bf16(false, (a), false, (b), (short)0,    \
                                          (c), false, false)

// ----------------------------- kernels -------------------------------------

__global__ void k_deg(const int* __restrict__ src, const int* __restrict__ dst,
                      float* deg_in, int* deg_out) {
  int e = blockIdx.x * blockDim.x + threadIdx.x;
  if (e >= cE) return;
  atomicAdd(&deg_in[dst[e]], 1.0f);
  atomicAdd(&deg_out[src[e]], 1);
}

__global__ void k_dinv(const float* __restrict__ deg_in, float* dinv) {
  int i = blockIdx.x * blockDim.x + threadIdx.x;
  if (i >= cN) return;
  dinv[i] = rsqrtf(deg_in[i] + 1.0f);
}

// h = x @ W1 (N x 128 @ 128 x 128). W1 staged to LDS (TDM when available).
// Block: 256 threads = 8 waves; wave = (row tile 0..1) x (32-col pair 0..3).
__global__ void __launch_bounds__(256) k_gemm_h(const float* __restrict__ x,
                                                const float* __restrict__ W1,
                                                float* __restrict__ h) {
  __shared__ float sW[cIN * cHID];  // 64 KB of the 320 KB WGP LDS
  int lane = threadIdx.x & 31, wave = threadIdx.x >> 5;

#if CDNA5_HAS_TDM
  if (wave == 0) {  // wave-uniform branch; EXEC all-ones inside
    unsigned long long ga = (unsigned long long)(uintptr_t)W1;
    unsigned lds = (unsigned)(uintptr_t)(void*)sW;  // addr[31:0] = LDS offset
    u32x4 g0;
    g0[0] = 1u;                                   // count=1, user descriptor
    g0[1] = lds;                                  // lds_addr
    g0[2] = (unsigned)(ga & 0xFFFFFFFFull);       // global_addr[31:0]
    g0[3] = (unsigned)((ga >> 32) & 0x1FFFFFFull) // global_addr[56:32]
            | (2u << 30);                         // type=2 ("image")
    i32x8 g1;
    g1[0] = (int)(2u << 16);       // workgroup_mask=0, data_size=2 (4 bytes)
    g1[1] = (int)(128u << 16);     // tensor_dim0[15:0] at bits[63:48]
    g1[2] = (int)(128u << 16);     // tensor_dim0[31:16]=0 | tensor_dim1[15:0]
    g1[3] = (int)(128u << 16);     // tensor_dim1[31:16]=0 | tile_dim0=128
    g1[4] = 128;                   // tile_dim1=128, tile_dim2=0
    g1[5] = 128;                   // tensor_dim0_stride[31:0] = 128
    g1[6] = 0;                     // stride0[47:32] | stride1[15:0]
    g1[7] = 0;
    i32x4 gz = {0, 0, 0, 0};
#if CDNA5_TDM_6ARG
    i32x8 gz8 = {0, 0, 0, 0, 0, 0, 0, 0};
    __builtin_amdgcn_tensor_load_to_lds(g0, g1, gz, gz, gz8, 0);
#else
    __builtin_amdgcn_tensor_load_to_lds(g0, g1, gz, gz, 0);
#endif
    __builtin_amdgcn_s_wait_tensorcnt(0);
  }
#else
  for (int i = threadIdx.x; i < cIN * cHID; i += 256) sW[i] = W1[i];
#endif
  __syncthreads();

  int half = lane >> 4, q = lane & 15;
  int m0 = blockIdx.x * 32 + (wave >> 2) * 16;
  int n0 = (wave & 3) * 32;
  f32x8 acc0 = {0.f, 0.f, 0.f, 0.f, 0.f, 0.f, 0.f, 0.f};
  f32x8 acc1 = acc0;
  for (int kk = 0; kk < cIN; kk += 32) {
    bf16x16 a  = load_a_frag(x, cIN, 1, m0, kk, half, q);
    bf16x16 b0 = load_b_frag(sW, cHID, 1, n0, kk, half, q);
    bf16x16 b1 = load_b_frag(sW, cHID, 1, n0 + 16, kk, half, q);
    acc0 = WMMA_BF16(a, b0, acc0);
    acc1 = WMMA_BF16(a, b1, acc1);
  }
#pragma unroll
  for (int r = 0; r < 8; ++r) {
    int row = m0 + (half << 3) + r;
    h[(size_t)row * cHID + n0 + q]      = acc0[r];
    h[(size_t)row * cHID + n0 + 16 + q] = acc1[r];
  }
}

__global__ void k_selfloop(const float* __restrict__ h,
                           const float* __restrict__ dinv, float* x1) {
  int idx = blockIdx.x * blockDim.x + threadIdx.x;
  if (idx >= cN * cHID) return;
  int i = idx >> 7;
  float d = dinv[i];
  x1[idx] = h[idx] * d * d;
}

// one wave per edge: x1[dst] += h[src] * dinv[src]*dinv[dst]
__global__ void k_agg(const int* __restrict__ src, const int* __restrict__ dst,
                      const float* __restrict__ h, const float* __restrict__ dinv,
                      float* x1) {
  int gw = (blockIdx.x * blockDim.x + threadIdx.x) >> 5;
  if (gw >= cE) return;
  int lane = threadIdx.x & 31;
  int s = src[gw], d = dst[gw];
  __builtin_prefetch(h + (size_t)s * cHID, 0, 0);   // global_prefetch_b8
  float nrm = dinv[s] * dinv[d];
  float4 hv = ((const float4*)(h + (size_t)s * cHID))[lane];
  float* p = x1 + (size_t)d * cHID + lane * 4;
  atomicAdd(p + 0, hv.x * nrm);
  atomicAdd(p + 1, hv.y * nrm);
  atomicAdd(p + 2, hv.z * nrm);
  atomicAdd(p + 3, hv.w * nrm);
}

// relu(x1+b1) in place; raw = x1 . w_score; keys = orderable(raw)
__global__ void k_post(float* x1, const float* __restrict__ b1,
                       const float* __restrict__ w_score, float* raw,
                       unsigned* keys) {
  int gw = (blockIdx.x * blockDim.x + threadIdx.x) >> 5;
  if (gw >= cN) return;
  int lane = threadIdx.x & 31;
  float4 v  = ((float4*)(x1 + (size_t)gw * cHID))[lane];
  float4 bb = ((const float4*)b1)[lane];
  v.x = fmaxf(v.x + bb.x, 0.f); v.y = fmaxf(v.y + bb.y, 0.f);
  v.z = fmaxf(v.z + bb.z, 0.f); v.w = fmaxf(v.w + bb.w, 0.f);
  ((float4*)(x1 + (size_t)gw * cHID))[lane] = v;
  float4 w = ((const float4*)w_score)[lane];
  float s = v.x * w.x + v.y * w.y + v.z * w.z + v.w * w.w;
#pragma unroll
  for (int off = 16; off > 0; off >>= 1) s += __shfl_xor(s, off, 32);
  if (lane == 0) { raw[gw] = s; keys[gw] = ordkey(s); }
}

__global__ void k_ctl_init(Ctl* c) {
  int t = threadIdx.x;
  if (t < 256) c->hist[t] = 0;
  if (t == 0) {
    c->prefix = 0; c->mask = 0; c->need = cK; c->T = 0;
    c->gtCount = 0; c->tieCount = 0; c->keptCount = 0; c->jmax = 0;
  }
}

__global__ void k_hist(Ctl* c, const unsigned* __restrict__ keys, int shift) {
  int i = blockIdx.x * blockDim.x + threadIdx.x;
  if (i >= cN) return;
  unsigned k = keys[i];
  if ((k & c->mask) == c->prefix)
    atomicAdd(&c->hist[(k >> shift) & 255u], 1u);
}

__global__ void k_scan(Ctl* c, int shift) {
  unsigned cum = 0; int b = 0;
  for (int i = 255; i >= 0; --i) {
    unsigned h = c->hist[i];
    if (cum + h >= c->need) { b = i; break; }
    cum += h;
  }
  c->prefix |= ((unsigned)b) << shift;
  c->mask   |= 0xFFu << shift;
  c->need   -= cum;
  c->T = c->prefix;
  for (int i = 0; i < 256; ++i) c->hist[i] = 0;
}

__global__ void k_gather(Ctl* c, const unsigned* __restrict__ keys,
                         int* kept_unsorted, int* tieBuf) {
  int i = blockIdx.x * blockDim.x + threadIdx.x;
  if (i >= cN) return;
  unsigned k = keys[i], T = c->T;
  if (k > T) {
    unsigned p = atomicAdd(&c->gtCount, 1u);
    if (p < (unsigned)cK) kept_unsorted[p] = i;
  } else if (k == T) {
    unsigned p = atomicAdd(&c->tieCount, 1u);
    if (p < 1024u) tieBuf[p] = i;
  }
}

// resolve ties: smallest indices first (single 1024-thread block bitonic)
__global__ void k_ties(Ctl* c, const int* __restrict__ tieBuf,
                       int* kept_unsorted) {
  __shared__ int si[1024];
  int t = threadIdx.x;
  unsigned tc = min(c->tieCount, 1024u);
  si[t] = (t < (int)tc) ? tieBuf[t] : 0x7FFFFFFF;
  for (unsigned size = 2; size <= 1024; size <<= 1)
    for (unsigned stride = size >> 1; stride > 0; stride >>= 1) {
      __syncthreads();
      unsigned j = t ^ stride;
      if (j > (unsigned)t) {
        bool up = ((t & size) == 0);
        int a = si[t], b = si[j];
        bool sw = up ? (a > b) : (a < b);   // ascending
        if (sw) { si[t] = b; si[j] = a; }
      }
    }
  __syncthreads();
  unsigned g = c->gtCount;
  unsigned needTies = (g < (unsigned)cK) ? ((unsigned)cK - g) : 0u;
  if ((unsigned)t < needTies) kept_unsorted[g + t] = si[t];
  if (t == 0) c->keptCount = cK;
}

// sort kept by (value desc, index asc); assign ranks; max-degree fallback
__global__ void k_sortkept(Ctl* c, const unsigned* __restrict__ keys,
                           const int* __restrict__ deg_out,
                           const int* __restrict__ kept_unsorted, int* kept,
                           int* keep_mask, int* cid_of_node) {
  __shared__ unsigned long long sk[1024];
  __shared__ int si[1024];
  __shared__ unsigned long long rk[1024];
  int t = threadIdx.x;
  int node = kept_unsorted[t];
  sk[t] = ((unsigned long long)keys[node] << 32) |
          (unsigned long long)(0xFFFFFFFFu - (unsigned)node);
  si[t] = node;
  for (unsigned size = 2; size <= 1024; size <<= 1)
    for (unsigned stride = size >> 1; stride > 0; stride >>= 1) {
      __syncthreads();
      unsigned j = t ^ stride;
      if (j > (unsigned)t) {
        bool up = ((t & size) == 0);
        unsigned long long a = sk[t], b = sk[j];
        bool sw = up ? (a < b) : (a > b);   // descending
        if (sw) {
          sk[t] = b; sk[j] = a;
          int tmp = si[t]; si[t] = si[j]; si[j] = tmp;
        }
      }
    }
  __syncthreads();
  node = si[t];
  kept[t] = node;
  keep_mask[node] = 1;
  cid_of_node[node] = t;
  rk[t] = ((unsigned long long)(unsigned)deg_out[node] << 32) |
          (unsigned long long)(1023 - t);
  __syncthreads();
  for (int s = 512; s > 0; s >>= 1) {
    if (t < s) rk[t] = (rk[t] >= rk[t + s]) ? rk[t] : rk[t + s];
    __syncthreads();
  }
  if (t == 0) c->jmax = 1023u - (unsigned)(rk[0] & 0xFFFFFFFFu);
}

// undirected neighbor scan: segment_max of encoded key via u64 atomicMax
__global__ void k_best(const int* __restrict__ src, const int* __restrict__ dst,
                       const int* __restrict__ keep_mask,
                       const int* __restrict__ deg_out,
                       unsigned long long* best) {
  int i = blockIdx.x * blockDim.x + threadIdx.x;
  if (i >= 2 * cE) return;
  int u, v;
  if (i < cE) { u = src[i]; v = dst[i]; }
  else        { u = dst[i - cE]; v = src[i - cE]; }
  if (!keep_mask[v]) return;
  unsigned long long key =
      (unsigned long long)((long long)deg_out[v] * (long long)cN +
                           (long long)(cN - 1 - v)) + 1ull;
  atomicMax(&best[u], key);
}

__global__ void k_cid(const Ctl* c, const int* __restrict__ keep_mask,
                      const int* __restrict__ cid_of_node,
                      const unsigned long long* __restrict__ best, int* cid) {
  int i = blockIdx.x * blockDim.x + threadIdx.x;
  if (i >= cN) return;
  if (keep_mask[i]) { cid[i] = cid_of_node[i]; return; }
  unsigned long long b = best[i];
  if (b > 0ull) {
    long long kv = (long long)(b - 1ull);
    int chosen = (cN - 1) - (int)(kv % (long long)cN);
    cid[i] = cid_of_node[chosen];
  } else {
    cid[i] = (int)c->jmax;
  }
}

// gated pooling: sums[cid] += x1 * tanh(raw), counts[cid] += 1
__global__ void k_pool(const float* __restrict__ x1,
                       const float* __restrict__ raw,
                       const int* __restrict__ cid, float* sums, float* counts) {
  int gw = (blockIdx.x * blockDim.x + threadIdx.x) >> 5;
  if (gw >= cN) return;
  int lane = threadIdx.x & 31;
  float g = tanhf(raw[gw]);
  int cc = cid[gw];
  float4 v = ((const float4*)(x1 + (size_t)gw * cHID))[lane];
  float* p = sums + (size_t)cc * cHID + lane * 4;
  atomicAdd(p + 0, v.x * g);
  atomicAdd(p + 1, v.y * g);
  atomicAdd(p + 2, v.z * g);
  atomicAdd(p + 3, v.w * g);
  if (lane == 0) atomicAdd(&counts[cc], 1.0f);
}

__global__ void k_xp(const float* __restrict__ sums,
                     const float* __restrict__ counts, float* x_p) {
  int idx = blockIdx.x * blockDim.x + threadIdx.x;
  if (idx >= cK * cHID) return;
  x_p[idx] = sums[idx] / fmaxf(counts[idx >> 7], 1.0f);
}

__global__ void k_adj(const int* __restrict__ src, const int* __restrict__ dst,
                      const int* __restrict__ cid, float* A) {
  int e = blockIdx.x * blockDim.x + threadIdx.x;
  if (e >= cE) return;
  int cu = cid[src[e]], cv = cid[dst[e]];
  if (cu != cv) A[(size_t)cu * cK + cv] = 1.0f;
}

__global__ void k_diag(float* A) {
  int cc = blockIdx.x * blockDim.x + threadIdx.x;
  if (cc < cK) A[(size_t)cc * cK + cc] = 1.0f;
}

// dpi[v] = rsqrt(sum_u Ahat[u][v]); one wave per column
__global__ void k_dpi(const float* __restrict__ A, float* dpi) {
  int gw = (blockIdx.x * blockDim.x + threadIdx.x) >> 5;
  if (gw >= cK) return;
  int lane = threadIdx.x & 31;
  float s = 0.f;
  for (int u = lane; u < cK; u += 32) s += A[(size_t)u * cK + gw];
#pragma unroll
  for (int off = 16; off > 0; off >>= 1) s += __shfl_xor(s, off, 32);
  if (lane == 0) dpi[gw] = rsqrtf(s);
}

__global__ void k_norm(float* A, const float* __restrict__ dpi) {
  int idx = blockIdx.x * blockDim.x + threadIdx.x;
  if (idx >= cK * cK) return;
  int u = idx >> 10, v = idx & 1023;
  A[idx] *= dpi[u] * dpi[v];
}

// y = x_p @ W2 (1024x128 @ 128x64); 2 waves x 2 col tiles
__global__ void __launch_bounds__(64) k_gemm_y(const float* __restrict__ x_p,
                                               const float* __restrict__ W2,
                                               float* y) {
  int lane = threadIdx.x & 31, wave = threadIdx.x >> 5;
  int half = lane >> 4, q = lane & 15;
  int m0 = blockIdx.x * 16, n0 = wave * 32;
  f32x8 acc0 = {0.f, 0.f, 0.f, 0.f, 0.f, 0.f, 0.f, 0.f};
  f32x8 acc1 = acc0;
  for (int kk = 0; kk < cHID; kk += 32) {
    bf16x16 a  = load_a_frag(x_p, cHID, 1, m0, kk, half, q);
    bf16x16 b0 = load_b_frag(W2, cOUT, 1, n0, kk, half, q);
    bf16x16 b1 = load_b_frag(W2, cOUT, 1, n0 + 16, kk, half, q);
    acc0 = WMMA_BF16(a, b0, acc0);
    acc1 = WMMA_BF16(a, b1, acc1);
  }
#pragma unroll
  for (int r = 0; r < 8; ++r) {
    int row = m0 + (half << 3) + r;
    y[(size_t)row * cOUT + n0 + q]      = acc0[r];
    y[(size_t)row * cOUT + n0 + 16 + q] = acc1[r];
  }
}

// xpo = normA^T @ y + b2  (A element (m,k) = normA[k*cK + m])
__global__ void __launch_bounds__(64) k_gemm_z(const float* __restrict__ normA,
                                               const float* __restrict__ y,
                                               const float* __restrict__ b2,
                                               float* xpo) {
  int lane = threadIdx.x & 31, wave = threadIdx.x >> 5;
  int half = lane >> 4, q = lane & 15;
  int m0 = blockIdx.x * 16, n0 = wave * 32;
  f32x8 acc0 = {0.f, 0.f, 0.f, 0.f, 0.f, 0.f, 0.f, 0.f};
  f32x8 acc1 = acc0;
  for (int kk = 0; kk < cK; kk += 32) {
    bf16x16 a  = load_a_frag(normA, 1, cK, m0, kk, half, q);
    bf16x16 b0 = load_b_frag(y, cOUT, 1, n0, kk, half, q);
    bf16x16 b1 = load_b_frag(y, cOUT, 1, n0 + 16, kk, half, q);
    acc0 = WMMA_BF16(a, b0, acc0);
    acc1 = WMMA_BF16(a, b1, acc1);
  }
  float bias0 = b2[n0 + q], bias1 = b2[n0 + 16 + q];
#pragma unroll
  for (int r = 0; r < 8; ++r) {
    int row = m0 + (half << 3) + r;
    xpo[(size_t)row * cOUT + n0 + q]      = acc0[r] + bias0;
    xpo[(size_t)row * cOUT + n0 + 16 + q] = acc1[r] + bias1;
  }
}

// logits = x1 @ Wskip + bskip + xpo[cid]
__global__ void __launch_bounds__(64) k_gemm_out(const float* __restrict__ x1,
                                                 const float* __restrict__ Wskip,
                                                 const float* __restrict__ bskip,
                                                 const float* __restrict__ xpo,
                                                 const int* __restrict__ cid,
                                                 float* out) {
  int lane = threadIdx.x & 31, wave = threadIdx.x >> 5;
  int half = lane >> 4, q = lane & 15;
  int m0 = blockIdx.x * 16, n0 = wave * 32;
  f32x8 acc0 = {0.f, 0.f, 0.f, 0.f, 0.f, 0.f, 0.f, 0.f};
  f32x8 acc1 = acc0;
  for (int kk = 0; kk < cHID; kk += 32) {
    bf16x16 a  = load_a_frag(x1, cHID, 1, m0, kk, half, q);
    bf16x16 b0 = load_b_frag(Wskip, cOUT, 1, n0, kk, half, q);
    bf16x16 b1 = load_b_frag(Wskip, cOUT, 1, n0 + 16, kk, half, q);
    acc0 = WMMA_BF16(a, b0, acc0);
    acc1 = WMMA_BF16(a, b1, acc1);
  }
  float bias0 = bskip[n0 + q], bias1 = bskip[n0 + 16 + q];
#pragma unroll
  for (int r = 0; r < 8; ++r) {
    int row = m0 + (half << 3) + r;
    int cc = cid[row];
    out[(size_t)row * cOUT + n0 + q] =
        acc0[r] + bias0 + xpo[(size_t)cc * cOUT + n0 + q];
    out[(size_t)row * cOUT + n0 + 16 + q] =
        acc1[r] + bias1 + xpo[(size_t)cc * cOUT + n0 + 16 + q];
  }
}

// ----------------------------- launch --------------------------------------

extern "C" void kernel_launch(void* const* d_in, const int* in_sizes, int n_in,
                              void* d_out, int out_size, void* d_ws,
                              size_t ws_size, hipStream_t stream) {
  const float* x       = (const float*)d_in[0];
  const int*   ei      = (const int*)d_in[1];
  const float* W1      = (const float*)d_in[2];
  const float* b1      = (const float*)d_in[3];
  const float* w_score = (const float*)d_in[4];
  const float* W2      = (const float*)d_in[5];
  const float* b2      = (const float*)d_in[6];
  const float* Wskip   = (const float*)d_in[7];
  const float* bskip   = (const float*)d_in[8];
  float* out = (float*)d_out;

  const int* src = ei;
  const int* dst = ei + cE;

  char* p = (char*)d_ws;
  auto carve = [&](size_t bytes) {
    char* r = p;
    p += (bytes + 255) & ~(size_t)255;
    return r;
  };
  Ctl*   ctl        = (Ctl*)carve(sizeof(Ctl));
  float* h          = (float*)carve((size_t)cN * cHID * 4);
  float* x1         = (float*)carve((size_t)cN * cHID * 4);
  float* deg_in     = (float*)carve((size_t)cN * 4);
  float* dinv       = (float*)carve((size_t)cN * 4);
  int*   deg_out    = (int*)carve((size_t)cN * 4);
  float* raw        = (float*)carve((size_t)cN * 4);
  unsigned* keys    = (unsigned*)carve((size_t)cN * 4);
  unsigned long long* best = (unsigned long long*)carve((size_t)cN * 8);
  int*   cid        = (int*)carve((size_t)cN * 4);
  int*   cid_of     = (int*)carve((size_t)cN * 4);
  int*   keep_mask  = (int*)carve((size_t)cN * 4);
  int*   kept_uns   = (int*)carve((size_t)cK * 4);
  int*   tieBuf     = (int*)carve(1024 * 4);
  int*   kept       = (int*)carve((size_t)cK * 4);
  float* sums       = (float*)carve((size_t)cK * cHID * 4);
  float* counts     = (float*)carve((size_t)cK * 4);
  float* x_p        = (float*)carve((size_t)cK * cHID * 4);
  float* Amat       = (float*)carve((size_t)cK * cK * 4);
  float* dpi        = (float*)carve((size_t)cK * 4);
  float* y          = (float*)carve((size_t)cK * cOUT * 4);
  float* xpo        = (float*)carve((size_t)cK * cOUT * 4);

  // zero accumulation buffers (captured in graph, replayed every call)
  hipMemsetAsync(deg_in, 0, (size_t)cN * 4, stream);
  hipMemsetAsync(deg_out, 0, (size_t)cN * 4, stream);
  hipMemsetAsync(keep_mask, 0, (size_t)cN * 4, stream);
  hipMemsetAsync(best, 0, (size_t)cN * 8, stream);
  hipMemsetAsync(sums, 0, (size_t)cK * cHID * 4, stream);
  hipMemsetAsync(counts, 0, (size_t)cK * 4, stream);
  hipMemsetAsync(Amat, 0, (size_t)cK * cK * 4, stream);

  // degrees + normalization
  k_deg<<<NB(cE, 256), 256, 0, stream>>>(src, dst, deg_in, deg_out);
  k_dinv<<<NB(cN, 256), 256, 0, stream>>>(deg_in, dinv);

  // h = x @ W1 (WMMA + TDM-staged W1; 32 rows x 128 cols per block)
  k_gemm_h<<<cN / 32, 256, 0, stream>>>(x, W1, h);

  // GCN aggregate
  k_selfloop<<<NB(cN * cHID, 256), 256, 0, stream>>>(h, dinv, x1);
  k_agg<<<NB(cE * 32, 256), 256, 0, stream>>>(src, dst, h, dinv, x1);

  // relu + bias + score
  k_post<<<NB(cN * 32, 256), 256, 0, stream>>>(x1, b1, w_score, raw, keys);

  // exact top-K via 4-pass radix select
  k_ctl_init<<<1, 256, 0, stream>>>(ctl);
  for (int pshift = 24; pshift >= 0; pshift -= 8) {
    k_hist<<<NB(cN, 256), 256, 0, stream>>>(ctl, keys, pshift);
    k_scan<<<1, 1, 0, stream>>>(ctl, pshift);
  }
  k_gather<<<NB(cN, 256), 256, 0, stream>>>(ctl, keys, kept_uns, tieBuf);
  k_ties<<<1, 1024, 0, stream>>>(ctl, tieBuf, kept_uns);
  k_sortkept<<<1, 1024, 0, stream>>>(ctl, keys, deg_out, kept_uns, kept,
                                     keep_mask, cid_of);

  // neighbor scan + cluster ids
  k_best<<<NB(2 * cE, 256), 256, 0, stream>>>(src, dst, keep_mask, deg_out, best);
  k_cid<<<NB(cN, 256), 256, 0, stream>>>(ctl, keep_mask, cid_of, best, cid);

  // gated mean pooling
  k_pool<<<NB(cN * 32, 256), 256, 0, stream>>>(x1, raw, cid, sums, counts);
  k_xp<<<NB(cK * cHID, 256), 256, 0, stream>>>(sums, counts, x_p);

  // pooled adjacency + symmetric normalization
  k_adj<<<NB(cE, 256), 256, 0, stream>>>(src, dst, cid, Amat);
  k_diag<<<NB(cK, 256), 256, 0, stream>>>(Amat);
  k_dpi<<<NB(cK * 32, 256), 256, 0, stream>>>(Amat, dpi);
  k_norm<<<NB(cK * cK, 256), 256, 0, stream>>>(Amat, dpi);

  // pooled GEMMs (WMMA, 2 tiles per wave)
  k_gemm_y<<<cK / 16, 64, 0, stream>>>(x_p, W2, y);
  k_gemm_z<<<cK / 16, 64, 0, stream>>>(Amat, y, b2, xpo);

  // final fused GEMM + broadcast + skip (WMMA, 2 tiles per wave)
  k_gemm_out<<<cN / 16, 64, 0, stream>>>(x1, Wskip, bskip, xpo, cid, out);
}
